// LSTM_46883863003345
// MI455X (gfx1250) — matrix operand order repaired and checked by reading, not compiled, not run on hardware
//
#include <hip/hip_runtime.h>
#include <hip/hip_bf16.h>
#include <stdint.h>

// ---------------- types for CDNA5 WMMA (wave32) ----------------
typedef __attribute__((ext_vector_type(16))) __bf16 v16bf;
typedef __attribute__((ext_vector_type(8)))  __bf16 v8bf;
typedef __attribute__((ext_vector_type(8)))  float  v8f;

#define SEQ   512
#define BATCH 64
#define IN_D  256
#define HID   512
#define GCOLS 2048   // 4 gates * HID

__device__ __forceinline__ uint16_t f32_to_bf16_rne(float f) {
    uint32_t u = __float_as_uint(f);
    return (uint16_t)((u + 0x7FFFu + ((u >> 16) & 1u)) >> 16);
}

__device__ __forceinline__ float sigmoidf_fast(float x) {
    return 1.0f / (1.0f + __expf(-x));
}

// ---------------- elementwise helpers ----------------
__global__ void cvt_f32_to_bf16(const float* __restrict__ src,
                                uint16_t* __restrict__ dst, int n) {
    int i = blockIdx.x * blockDim.x + threadIdx.x;
    if (i < n) dst[i] = f32_to_bf16_rne(src[i]);
}

__global__ void bias_combine(const float* __restrict__ wf, const float* __restrict__ uf,
                             const float* __restrict__ wi, const float* __restrict__ ui,
                             const float* __restrict__ wo, const float* __restrict__ uo,
                             const float* __restrict__ wc, const float* __restrict__ uc,
                             float* __restrict__ out) {
    int i = blockIdx.x * blockDim.x + threadIdx.x;
    if (i >= 4 * HID) return;
    int g = i / HID, n = i % HID;
    const float* W[4] = {wf, wi, wo, wc};
    const float* U[4] = {uf, ui, uo, uc};
    out[i] = W[g][n] + U[g][n];
}

__global__ void init_state(const float* __restrict__ h0, const float* __restrict__ c0,
                           uint16_t* __restrict__ hbf, float* __restrict__ c_state) {
    int i = blockIdx.x * blockDim.x + threadIdx.x;
    if (i < BATCH * HID) {
        hbf[i] = f32_to_bf16_rne(h0[i]);
        c_state[i] = c0[i];
    }
}

__global__ void finalize(const float* __restrict__ h_last, const float* __restrict__ c_state,
                         float* __restrict__ out_h, float* __restrict__ out_c) {
    int i = blockIdx.x * blockDim.x + threadIdx.x;
    if (i < BATCH * HID) {
        out_h[i] = h_last[i];
        out_c[i] = c_state[i];
    }
}

// ---------------- fragment loaders (per ISA 05_wmma layouts) ----------------
// A (16x32 bf16): lane l (m = l&15): elems 0..7 = A[m][kb + (l>>4)*8 + j],
//                 elems 8..15       = A[m][kb + 16 + (l>>4)*8 + j]
__device__ __forceinline__ v16bf load_a_frag(const uint16_t* __restrict__ arow,
                                             int kb, int hi) {
    v8bf lo = *reinterpret_cast<const v8bf*>(arow + kb + hi * 8);
    v8bf hi8 = *reinterpret_cast<const v8bf*>(arow + kb + 16 + hi * 8);
    return __builtin_shufflevector(lo, hi8, 0, 1, 2, 3, 4, 5, 6, 7,
                                   8, 9, 10, 11, 12, 13, 14, 15);
}
// B (32x16 bf16): lane l (n = l&15): elems 0..7 = B[kb + (l>>4)*8 + j][n],
//                 elems 8..15 = B[kb + 16 + (l>>4)*8 + j][n]  (B[k][n] = U[n][k])
__device__ __forceinline__ v16bf load_b_frag(const uint16_t* __restrict__ brow,
                                             int kb, int hi) {
    v8bf lo = *reinterpret_cast<const v8bf*>(brow + kb + hi * 8);
    v8bf hi8 = *reinterpret_cast<const v8bf*>(brow + kb + 16 + hi * 8);
    return __builtin_shufflevector(lo, hi8, 0, 1, 2, 3, 4, 5, 6, 7,
                                   8, 9, 10, 11, 12, 13, 14, 15);
}

// ---------------- phase 1: xg[m, g*512+n] = x @ W_g^T ----------------
// M=32768 (2048 m-tiles), N=2048 (128 n-tiles), K=256. One wave per 16x16 tile.
__global__ void __launch_bounds__(256)
xproj_wmma(const uint16_t* __restrict__ xbf,   // (32768, 256) bf16
           const uint16_t* __restrict__ wbf,   // (4*512, 256) bf16, gate-major rows
           float* __restrict__ xg) {           // (32768, 2048) f32
    const int lane = threadIdx.x & 31;
    const int wave = blockIdx.x * (blockDim.x >> 5) + (threadIdx.x >> 5);
    const int NT = GCOLS / 16;                 // 128
    const int mt = wave / NT;
    const int nt = wave % NT;
    const int l16 = lane & 15;
    const int hi  = lane >> 4;

    const uint16_t* arow = xbf + (size_t)(mt * 16 + l16) * IN_D;
    const uint16_t* brow = wbf + (size_t)(nt * 16 + l16) * IN_D;

    v8f acc = {};
#pragma unroll
    for (int kb = 0; kb < IN_D; kb += 32) {
        v16bf a = load_a_frag(arow, kb, hi);
        v16bf b = load_b_frag(brow, kb, hi);
        acc = __builtin_amdgcn_wmma_f32_16x16x32_bf16(false, a, false, b,
                                                      (short)0, acc, false, false);
    }
    const int ncol = nt * 16 + l16;
    float* crow = xg + (size_t)(mt * 16 + hi * 8) * GCOLS + ncol;
#pragma unroll
    for (int v = 0; v < 8; ++v)
        crow[(size_t)v * GCOLS] = acc[v];
}

// ---------------- phase 2: one recurrence step ----------------
// 4 m-tiles x 32 n-tiles = 128 waves. Each wave owns a 16x16 (b,n) patch for
// ALL FOUR gates (64 WMMAs), then does the cell update locally.
__global__ void __launch_bounds__(256)
lstm_step(const uint16_t* __restrict__ hbf_in,  // (64,512) bf16
          uint16_t* __restrict__ hbf_out,       // (64,512) bf16
          float* __restrict__ c_state,          // (64,512) f32, in-place
          const uint16_t* __restrict__ ubf,     // (4*512, 512) bf16
          const float* __restrict__ xg_t,       // (64, 2048) f32 slice at t
          const float* __restrict__ bias,       // (4*512) f32  (Wb+Ub)
          float* __restrict__ hout) {           // (64,512) f32 -> h_seq[t]
    const int lane = threadIdx.x & 31;
    const int wave = blockIdx.x * (blockDim.x >> 5) + (threadIdx.x >> 5);
    const int nt = wave & 31;
    const int mt = wave >> 5;
    const int l16 = lane & 15;
    const int hi  = lane >> 4;
    const int n   = nt * 16 + l16;

    const uint16_t* arow = hbf_in + (size_t)(mt * 16 + l16) * HID;
    const uint16_t* bF = ubf + (size_t)(0 * HID + n) * HID;
    const uint16_t* bI = ubf + (size_t)(1 * HID + n) * HID;
    const uint16_t* bO = ubf + (size_t)(2 * HID + n) * HID;
    const uint16_t* bC = ubf + (size_t)(3 * HID + n) * HID;

    v8f accf = {}, acci = {}, acco = {}, accc = {};
#pragma unroll 4
    for (int kb = 0; kb < HID; kb += 32) {
        v16bf a = load_a_frag(arow, kb, hi);
        v16bf f = load_b_frag(bF, kb, hi);
        accf = __builtin_amdgcn_wmma_f32_16x16x32_bf16(false, a, false, f,
                                                       (short)0, accf, false, false);
        v16bf i = load_b_frag(bI, kb, hi);
        acci = __builtin_amdgcn_wmma_f32_16x16x32_bf16(false, a, false, i,
                                                       (short)0, acci, false, false);
        v16bf o = load_b_frag(bO, kb, hi);
        acco = __builtin_amdgcn_wmma_f32_16x16x32_bf16(false, a, false, o,
                                                       (short)0, acco, false, false);
        v16bf c = load_b_frag(bC, kb, hi);
        accc = __builtin_amdgcn_wmma_f32_16x16x32_bf16(false, a, false, c,
                                                       (short)0, accc, false, false);
    }

    const float biasF = bias[0 * HID + n];
    const float biasI = bias[1 * HID + n];
    const float biasO = bias[2 * HID + n];
    const float biasC = bias[3 * HID + n];
#pragma unroll
    for (int v = 0; v < 8; ++v) {
        const int m = mt * 16 + hi * 8 + v;
        const size_t xrow = (size_t)m * GCOLS;
        float pf = accf[v] + xg_t[xrow + 0 * HID + n] + biasF;
        float pi = acci[v] + xg_t[xrow + 1 * HID + n] + biasI;
        float po = acco[v] + xg_t[xrow + 2 * HID + n] + biasO;
        float pc = accc[v] + xg_t[xrow + 3 * HID + n] + biasC;
        float f = sigmoidf_fast(pf);
        float i = sigmoidf_fast(pi);
        float o = sigmoidf_fast(po);
        float g = tanhf(pc);
        const size_t sidx = (size_t)m * HID + n;
        float cn = f * c_state[sidx] + i * g;   // (b,n) exclusively owned -> in-place ok
        c_state[sidx] = cn;
        float h = o * tanhf(cn);
        hout[sidx] = h;
        hbf_out[sidx] = f32_to_bf16_rne(h);
    }
}

// ---------------- host-side orchestration ----------------
extern "C" void kernel_launch(void* const* d_in, const int* in_sizes, int n_in,
                              void* d_out, int out_size, void* d_ws, size_t ws_size,
                              hipStream_t stream) {
    (void)in_sizes; (void)n_in; (void)out_size; (void)ws_size;
    char* ws = (char*)d_ws;
    size_t off = 0;
    auto carve = [&](size_t bytes) {
        void* p = ws + off;
        off += (bytes + 255) & ~(size_t)255;
        return p;
    };
    uint16_t* xbf     = (uint16_t*)carve((size_t)SEQ * BATCH * IN_D * 2);   // 16 MB
    uint16_t* wbf     = (uint16_t*)carve((size_t)4 * HID * IN_D * 2);       // 1 MB
    uint16_t* ubf     = (uint16_t*)carve((size_t)4 * HID * HID * 2);        // 2 MB
    float*    bias    = (float*)   carve((size_t)4 * HID * 4);
    uint16_t* hbf0    = (uint16_t*)carve((size_t)BATCH * HID * 2);
    uint16_t* hbf1    = (uint16_t*)carve((size_t)BATCH * HID * 2);
    float*    c_state = (float*)   carve((size_t)BATCH * HID * 4);
    float*    xg      = (float*)   carve((size_t)SEQ * BATCH * GCOLS * 4);  // 268 MB

    const float* x  = (const float*)d_in[0];
    const float* h0 = (const float*)d_in[1];
    const float* c0 = (const float*)d_in[2];
    // gate order f, i, o, c matches d_in layout
    const int widx[4] = {3, 7, 11, 15};
    const int wbidx[4] = {4, 8, 12, 16};
    const int uidx[4] = {5, 9, 13, 17};
    const int ubidx[4] = {6, 10, 14, 18};

    // ---- convert inputs to bf16 ----
    {
        int n = SEQ * BATCH * IN_D;
        cvt_f32_to_bf16<<<(n + 255) / 256, 256, 0, stream>>>(x, xbf, n);
    }
    for (int g = 0; g < 4; ++g) {
        int nw = HID * IN_D;
        cvt_f32_to_bf16<<<(nw + 255) / 256, 256, 0, stream>>>(
            (const float*)d_in[widx[g]], wbf + (size_t)g * nw, nw);
        int nu = HID * HID;
        cvt_f32_to_bf16<<<(nu + 255) / 256, 256, 0, stream>>>(
            (const float*)d_in[uidx[g]], ubf + (size_t)g * nu, nu);
    }
    bias_combine<<<(4 * HID + 255) / 256, 256, 0, stream>>>(
        (const float*)d_in[wbidx[0]], (const float*)d_in[ubidx[0]],
        (const float*)d_in[wbidx[1]], (const float*)d_in[ubidx[1]],
        (const float*)d_in[wbidx[2]], (const float*)d_in[ubidx[2]],
        (const float*)d_in[wbidx[3]], (const float*)d_in[ubidx[3]], bias);
    init_state<<<(BATCH * HID + 255) / 256, 256, 0, stream>>>(h0, c0, hbf0, c_state);

    // ---- phase 1: all input projections (one big WMMA GEMM) ----
    {
        // waves = (32768/16) * (2048/16) = 262144; 8 waves per block
        int blocks = (SEQ * BATCH / 16) * (GCOLS / 16) / 8;  // 32768
        xproj_wmma<<<blocks, 256, 0, stream>>>(xbf, wbf, xg);
    }

    // ---- phase 2: serial recurrence, stream-ordered steps ----
    float* out_seq = (float*)d_out;
    for (int t = 0; t < SEQ; ++t) {
        const uint16_t* hin = (t & 1) ? hbf1 : hbf0;
        uint16_t* hob       = (t & 1) ? hbf0 : hbf1;
        lstm_step<<<16, 256, 0, stream>>>(
            hin, hob, c_state, ubf,
            xg + (size_t)t * BATCH * GCOLS, bias,
            out_seq + (size_t)t * BATCH * HID);
    }

    // ---- append final (h, c) after h_seq ----
    finalize<<<(BATCH * HID + 255) / 256, 256, 0, stream>>>(
        out_seq + (size_t)(SEQ - 1) * BATCH * HID, c_state,
        out_seq + (size_t)SEQ * BATCH * HID,
        out_seq + (size_t)SEQ * BATCH * HID + BATCH * HID);
}
